// TSCMamba_SingleSensor_v1_63806034149886
// MI455X (gfx1250) — compile-verified, matrix operand order checked by reading, not run on hardware
//
#include <hip/hip_runtime.h>

typedef __attribute__((ext_vector_type(16))) _Float16 v16h;
typedef __attribute__((ext_vector_type(8)))  _Float16 h8;
typedef __attribute__((ext_vector_type(8)))  float    v8f;

#define BB   16
#define CC   64
#define IMG  224
#define PTCH 16
#define PS   256
#define HW   196
#define DST  16
#define NCLS 18

// ---------------------------------------------------------------------------
// device helpers
// ---------------------------------------------------------------------------
__device__ __forceinline__ float act_apply(float v, int act) {
  if (act == 1) return 0.5f * v * (1.0f + erff(v * 0.70710678118654752f));   // exact gelu
  if (act == 2) return v / (1.0f + expf(-v));                               // silu
  if (act == 3) return (v > 20.0f) ? v : log1pf(expf(v));                   // softplus
  return v;
}
__device__ __forceinline__ v16h cat16(h8 a, h8 b) {
  return __builtin_shufflevector(a, b, 0,1,2,3,4,5,6,7,8,9,10,11,12,13,14,15);
}

// ---------------------------------------------------------------------------
// weight convert: dst (Npad x Kpad) f16 row-major = B^T (N x K), zero padded.
// trans=1: src is (K x N) row-major; trans=0: src is (N x K) row-major.
// ---------------------------------------------------------------------------
__global__ void wconv_f16_k(const float* __restrict__ src, _Float16* __restrict__ dst,
                            int N, int K, int Npad, int Kpad, int trans) {
  int idx = blockIdx.x * blockDim.x + threadIdx.x;
  int total = Npad * Kpad;
  if (idx >= total) return;
  int n = idx / Kpad, k = idx % Kpad;
  float v = 0.0f;
  if (n < N && k < K) v = trans ? src[(size_t)k * N + n] : src[(size_t)n * K + k];
  dst[idx] = (_Float16)v;
}

// activation convert: src f32 (M x K, row stride lda) -> dst f16 (M x Kpad), zero pad
__global__ void acvt_k(const float* __restrict__ src, _Float16* __restrict__ dst,
                       int M, int K, int lda, int Kpad) {
  int idx = blockIdx.x * blockDim.x + threadIdx.x;
  if (idx >= M * Kpad) return;
  int m = idx / Kpad, k = idx % Kpad;
  dst[idx] = (k < K) ? (_Float16)src[(size_t)m * lda + k] : (_Float16)0.0f;
}

// ---------------------------------------------------------------------------
// Generic WMMA GEMM:  C(MxN,f32) = act( A * B^T + bias ) + res
// A  : f16 (M x Kpad)       row-major, K zero-padded to mult of 32
// Bt : f16 (Npad32 x Kpad)  row-major (B transposed), zero padded
// One wave -> 16x32 tile (A fragment reused over two N tiles, 2 WMMAs/iter).
// block = 128 threads = 4 waves; grid = (M/16, ceil(N/128)).
// ---------------------------------------------------------------------------
__global__ void gemm_wmma_k(const _Float16* __restrict__ A, const _Float16* __restrict__ Bt,
                            float* __restrict__ C, int M, int Kpad, int N, int ldc,
                            const float* __restrict__ bias,
                            const float* __restrict__ res, int act) {
  const int wave  = threadIdx.x >> 5;
  const int lane  = threadIdx.x & 31;
  const int mtile = blockIdx.x;
  const int nbase = (blockIdx.y * 4 + wave) * 32;
  if (nbase >= N) return;                    // wave-uniform
  const int r  = lane & 15;
  const int hi = lane >> 4;
  const _Float16* Arow  = A  + (size_t)(mtile * 16 + r) * Kpad;
  const _Float16* Brow0 = Bt + (size_t)(nbase + r) * Kpad;
  const _Float16* Brow1 = Bt + (size_t)(nbase + 16 + r) * Kpad;
  v8f acc0 = {}, acc1 = {};
  for (int k0 = 0; k0 < Kpad; k0 += 32) {
    const int ka = k0 + hi * 8;
    v16h af  = cat16(*(const h8*)(Arow  + ka), *(const h8*)(Arow  + ka + 16));
    v16h bf0 = cat16(*(const h8*)(Brow0 + ka), *(const h8*)(Brow0 + ka + 16));
    v16h bf1 = cat16(*(const h8*)(Brow1 + ka), *(const h8*)(Brow1 + ka + 16));
    acc0 = __builtin_amdgcn_wmma_f32_16x16x32_f16(false, af, false, bf0,
                                                  (short)0, acc0, false, false);
    acc1 = __builtin_amdgcn_wmma_f32_16x16x32_f16(false, af, false, bf1,
                                                  (short)0, acc1, false, false);
  }
#pragma unroll
  for (int t = 0; t < 2; ++t) {
    const int cn = nbase + t * 16 + r;
    if (cn < N) {
      v8f acc = t ? acc1 : acc0;
#pragma unroll
      for (int v = 0; v < 8; ++v) {
        int rm = mtile * 16 + v + hi * 8;
        float val = acc[v];
        if (bias) val += bias[cn];
        val = act_apply(val, act);
        if (res)  val += res[(size_t)rm * ldc + cn];
        C[(size_t)rm * ldc + cn] = val;
      }
    }
  }
}

// ---------------------------------------------------------------------------
// Patch-embed conv as implicit GEMM (stride-16 16x16 patches, VALID).
// M row = b*196 + patch, K = c_in*256 + py*16 + px, N = c_out (64).
// Output stored as xp[(b*64 + c_out)*196 + patch]  (== reshape(B,C,HW)).
// grid = 196 (M tiles), block = 128 (4 waves -> 64 c_out).
// ---------------------------------------------------------------------------
__global__ void patch_gemm_k(const float* __restrict__ x, const _Float16* __restrict__ wT,
                             const float* __restrict__ cb, float* __restrict__ out) {
  const int wave = threadIdx.x >> 5;
  const int lane = threadIdx.x & 31;
  const int n0   = wave * 16;
  const int r    = lane & 15;
  const int hi   = lane >> 4;
  const int row  = blockIdx.x * 16 + r;      // m index for A
  const int b    = row / HW, pidx = row % HW;
  const int ph   = pidx / 14, pw = pidx % 14;
  const float* xb = x + (size_t)b * CC * IMG * IMG + (size_t)(ph * PTCH) * IMG + pw * PTCH;
  const _Float16* wrow = wT + (size_t)(n0 + r) * (CC * PTCH * PTCH);
  v8f acc = {};
  for (int k0 = 0; k0 < CC * PTCH * PTCH; k0 += 32) {
    const int ka = k0 + hi * 8;
    v16h af, bf;
#pragma unroll
    for (int hf = 0; hf < 2; ++hf) {
      int kk = ka + hf * 16;
      int ci = kk >> 8, py = (kk >> 4) & 15, px = kk & 15;
      const float4* p4 = (const float4*)(xb + ((size_t)ci * IMG + py) * IMG + px);
      float4 a0 = p4[0], a1 = p4[1];
      af[hf * 8 + 0] = (_Float16)a0.x; af[hf * 8 + 1] = (_Float16)a0.y;
      af[hf * 8 + 2] = (_Float16)a0.z; af[hf * 8 + 3] = (_Float16)a0.w;
      af[hf * 8 + 4] = (_Float16)a1.x; af[hf * 8 + 5] = (_Float16)a1.y;
      af[hf * 8 + 6] = (_Float16)a1.z; af[hf * 8 + 7] = (_Float16)a1.w;
    }
    bf = cat16(*(const h8*)(wrow + ka), *(const h8*)(wrow + ka + 16));
    acc = __builtin_amdgcn_wmma_f32_16x16x32_f16(false, af, false, bf,
                                                 (short)0, acc, false, false);
  }
  const int c = n0 + r;
  const float bias = cb[c];
#pragma unroll
  for (int v = 0; v < 8; ++v) {
    int m = blockIdx.x * 16 + v + hi * 8;
    int bb = m / HW, pp = m % HW;
    out[((size_t)bb * CC + c) * HW + pp] = acc[v] + bias;
  }
}

// ---------------------------------------------------------------------------
// LayerNorm over last dim, one wave per row.
// ---------------------------------------------------------------------------
__global__ void layernorm_k(const float* __restrict__ x, float* __restrict__ o,
                            const float* __restrict__ g, const float* __restrict__ bt,
                            int rows, int D) {
  int row  = blockIdx.x * (blockDim.x >> 5) + (threadIdx.x >> 5);
  int lane = threadIdx.x & 31;
  if (row >= rows) return;
  const float* xr = x + (size_t)row * D;
  float s = 0.f, s2 = 0.f;
  for (int i = lane; i < D; i += 32) { float v = xr[i]; s += v; s2 += v * v; }
#pragma unroll
  for (int m = 16; m >= 1; m >>= 1) { s += __shfl_xor(s, m, 32); s2 += __shfl_xor(s2, m, 32); }
  float mu  = s / (float)D;
  float var = s2 / (float)D - mu * mu;
  float inv = rsqrtf(var + 1e-5f);
  float* orow = o + (size_t)row * D;
  for (int i = lane; i < D; i += 32) orow[i] = (xr[i] - mu) * inv * g[i] + bt[i];
}

// causal depthwise conv (DCONV=4) + SiLU; dual store f32 (scan) + f16 (GEMM A)
__global__ void dwconv_silu_k(const float* __restrict__ xz, const float* __restrict__ w,
                              const float* __restrict__ bias, float* __restrict__ xc,
                              _Float16* __restrict__ xch, int Bn, int L, int di, int stride) {
  int idx = blockIdx.x * blockDim.x + threadIdx.x;
  if (idx >= Bn * L * di) return;
  int ch = idx % di; int t = idx / di; int l = t % L; int b = t / L;
  const float* base = xz + (size_t)(b * L) * stride + ch;
  float acc = bias[ch];
#pragma unroll
  for (int j = 0; j < 4; ++j) {
    int ls = l - 3 + j;
    if (ls >= 0) acc += base[(size_t)ls * stride] * w[ch * 4 + j];
  }
  float s = acc / (1.0f + expf(-acc));
  xc[idx]  = s;
  xch[idx] = (_Float16)s;
}

// selective scan: one thread per (b, d); h state (DST=16) in registers.
__global__ void scan_k(const float* __restrict__ xc, const float* __restrict__ dtb,
                       const float* __restrict__ xdbl, int ldx, int dr,
                       const float* __restrict__ A_log, const float* __restrict__ Dp,
                       float* __restrict__ y, int Bn, int L, int di) {
  int idx = blockIdx.x * blockDim.x + threadIdx.x;
  if (idx >= Bn * di) return;
  int d = idx % di, b = idx / di;
  float Arow[DST], h[DST];
#pragma unroll
  for (int s = 0; s < DST; ++s) { Arow[s] = -expf(A_log[d * DST + s]); h[s] = 0.f; }
  float Dd = Dp[d];
  for (int l = 0; l < L; ++l) {
    size_t rb = (size_t)(b * L + l);
    float dt = dtb[rb * di + d];
    float xv = xc[rb * di + d];
    const float* xr = xdbl + rb * ldx;
    float dtx = dt * xv, yv = 0.f;
#pragma unroll
    for (int s = 0; s < DST; ++s) {
      h[s] = __expf(dt * Arow[s]) * h[s] + dtx * xr[dr + s];
      yv  += h[s] * xr[dr + DST + s];
    }
    y[rb * di + d] = yv + Dd * xv;
  }
}

// yh = f16( y * silu(z) );  z = xz[:, di + ch], row stride = 2*di
__global__ void gate_k(const float* __restrict__ y, const float* __restrict__ xz,
                       _Float16* __restrict__ yh, int n, int di, int stride) {
  int idx = blockIdx.x * blockDim.x + threadIdx.x;
  if (idx >= n) return;
  int ch = idx % di; size_t r = (size_t)(idx / di);
  float z = xz[r * stride + di + ch];
  yh[idx] = (_Float16)(y[idx] * z / (1.0f + expf(-z)));
}

__global__ void flip_k(const float* __restrict__ src, float* __restrict__ dst,
                       int Bn, int L, int D) {
  int idx = blockIdx.x * blockDim.x + threadIdx.x;
  if (idx >= Bn * L * D) return;
  int d = idx % D; int l = (idx / D) % L; int b = idx / (D * L);
  dst[idx] = src[((size_t)b * L + (L - 1 - l)) * D + d];
}

__global__ void transpose_k(const float* __restrict__ src, float* __restrict__ dst,
                            int Bn, int L, int D) {   // (B,L,D) -> (B,D,L)
  int idx = blockIdx.x * blockDim.x + threadIdx.x;
  if (idx >= Bn * L * D) return;
  int d = idx % D; int l = (idx / D) % L; int b = idx / (D * L);
  dst[((size_t)b * D + d) * L + l] = src[idx];
}

__global__ void add_k(const float* __restrict__ a, const float* __restrict__ b,
                      float* __restrict__ o, int n) {
  int idx = blockIdx.x * blockDim.x + threadIdx.x;
  if (idx < n) o[idx] = a[idx] + b[idx];
}

__global__ void meanadd_k(const float* __restrict__ a, const float* __restrict__ b,
                          float* __restrict__ o, int Bn, int L, int D) {
  int idx = blockIdx.x * blockDim.x + threadIdx.x;
  if (idx >= Bn * D) return;
  int f = idx % D, bb = idx / D;
  float s = 0.f;
  for (int l = 0; l < L; ++l) { size_t r = ((size_t)bb * L + l) * D + f; s += a[r] + b[r]; }
  o[idx] = s / (float)L;
}

// ---------------------------------------------------------------------------
// host side
// ---------------------------------------------------------------------------
struct MambaP {
  const float *in_proj, *conv_w, *conv_b, *x_proj, *dt_proj_w, *dt_proj_b, *A_log, *D, *out_proj;
};
struct MambaWs { _Float16 *inT, *xT, *dtT, *outT; };

static inline int ru32(int x) { return (x + 31) & ~31; }

static void launch_gemm(hipStream_t s, const _Float16* A, const _Float16* Bt, float* C,
                        int M, int Kpad, int N, int ldc,
                        const float* bias, const float* res, int act) {
  dim3 g(M / 16, (N + 127) / 128);
  gemm_wmma_k<<<g, 128, 0, s>>>(A, Bt, C, M, Kpad, N, ldc, bias, res, act);
}
static void launch_wconv(hipStream_t s, const float* src, _Float16* dst,
                         int N, int K, int Kpad, int trans) {
  int total = ru32(N) * Kpad;
  wconv_f16_k<<<(total + 255) / 256, 256, 0, s>>>(src, dst, N, K, ru32(N), Kpad, trans);
}
static void launch_acvt(hipStream_t s, const float* src, _Float16* dst,
                        int M, int K, int lda, int Kpad) {
  int total = M * Kpad;
  acvt_k<<<(total + 255) / 256, 256, 0, s>>>(src, dst, M, K, lda, Kpad);
}

extern "C" void kernel_launch(void* const* d_in, const int* in_sizes, int n_in,
                              void* d_out, int out_size, void* d_ws, size_t ws_size,
                              hipStream_t stream) {
  (void)in_sizes; (void)n_in; (void)out_size;
  const float* x_cwt  = (const float*)d_in[0];
  const float* conv_w = (const float*)d_in[1];
  const float* conv_b = (const float*)d_in[2];
  const float* proj_w = (const float*)d_in[3];
  const float* proj_b = (const float*)d_in[4];
  const float* ln_g   = (const float*)d_in[5];
  const float* ln_b   = (const float*)d_in[6];
  auto fillP = [&](int base) {
    MambaP p;
    p.in_proj   = (const float*)d_in[base + 0];
    p.conv_w    = (const float*)d_in[base + 1];
    p.conv_b    = (const float*)d_in[base + 2];
    p.x_proj    = (const float*)d_in[base + 3];
    p.dt_proj_w = (const float*)d_in[base + 4];
    p.dt_proj_b = (const float*)d_in[base + 5];
    p.A_log     = (const float*)d_in[base + 6];
    p.D         = (const float*)d_in[base + 7];
    p.out_proj  = (const float*)d_in[base + 8];
    return p;
  };
  MambaP m1[2] = { fillP(7),  fillP(16) };
  MambaP m2[2] = { fillP(25), fillP(34) };
  const float* cls_w1 = (const float*)d_in[43];
  const float* cls_b1 = (const float*)d_in[44];
  const float* cls_w2 = (const float*)d_in[45];
  const float* cls_b2 = (const float*)d_in[46];

  // ---- workspace bump allocator ----
  char* wsb = (char*)d_ws; size_t off = 0;
  auto alloc = [&](size_t bytes) { void* p = wsb + off; off = (off + bytes + 255) & ~(size_t)255; return p; };
  auto f32b  = [&](size_t n) { return (float*)alloc(n * 4); };
  auto f16b  = [&](size_t n) { return (_Float16*)alloc(n * 2); };

  const int D1 = 256, di1 = 512, dr1 = 16, L1 = 64,  ldx1 = dr1 + 2 * DST; // 48
  const int D2 = 64,  di2 = 128, dr2 = 4,  L2 = 256, ldx2 = dr2 + 2 * DST; // 36

  // f16 weights (Bt: Npad32 x Kpad32)
  _Float16* convwT = f16b((size_t)64 * 16384);
  _Float16* projT  = f16b((size_t)256 * 224);
  MambaWs w1[2], w2[2];
  for (int i = 0; i < 2; ++i) {
    w1[i].inT  = f16b((size_t)1024 * 256);
    w1[i].xT   = f16b((size_t)64 * 512);
    w1[i].dtT  = f16b((size_t)512 * 32);
    w1[i].outT = f16b((size_t)256 * 512);
    w2[i].inT  = f16b((size_t)256 * 64);
    w2[i].xT   = f16b((size_t)64 * 128);
    w2[i].dtT  = f16b((size_t)128 * 32);
    w2[i].outT = f16b((size_t)64 * 128);
  }
  _Float16* cw1T = f16b((size_t)128 * 256);
  _Float16* cw2T = f16b((size_t)32 * 128);

  // activations
  const size_t NE = (size_t)1024 * 256;               // common (B*L*D) size = 262144
  float* xp_c = f32b((size_t)1024 * 196);
  float* xg   = f32b(NE);
  float* xln  = f32b(NE);
  float* fbuf = f32b(NE);
  float* fbuf2= f32b(NE);
  float* rbuf[8]; for (int i = 0; i < 8; ++i) rbuf[i] = f32b(NE);
  float* x1   = f32b(NE);
  float* xpm  = f32b(NE);
  float* xpmf = f32b(NE);
  float* ta   = f32b(NE);
  float* tb   = f32b(NE);
  float* tb2  = f32b(NE);
  float* x2   = f32b(NE);
  float* x3   = f32b((size_t)16 * 256);
  float* t1   = f32b((size_t)16 * 128);
  // mamba f32 temps (max over both configs)
  float* xz   = f32b((size_t)1024 * 1024);
  float* xc   = f32b((size_t)1024 * 512);
  float* xdbl = f32b((size_t)4096 * 36);
  float* dtb  = f32b((size_t)1024 * 512);
  float* ybuf = f32b((size_t)1024 * 512);
  // f16 activation staging
  _Float16* xph  = f16b((size_t)1024 * 224);
  _Float16* ah_x = f16b(NE);                  // block input,  M x D
  _Float16* xch  = f16b((size_t)1024 * 512);  // M x di
  _Float16* dth  = f16b((size_t)4096 * 32);   // M x 32
  _Float16* yh   = f16b((size_t)1024 * 512);  // M x di
  _Float16* x3h  = f16b((size_t)16 * 256);
  _Float16* t1h  = f16b((size_t)16 * 128);
  if (off > ws_size) return;                  // workspace too small: bail safely

  // ---- stage 0: convert all weights to f16 B^T layout ----
  launch_wconv(stream, conv_w, convwT, 64, 16384, 16384, 0);
  launch_wconv(stream, proj_w, projT, 256, 196, 224, 1);
  for (int i = 0; i < 2; ++i) {
    launch_wconv(stream, m1[i].in_proj,   w1[i].inT,  2 * di1, D1,  D1,  1);
    launch_wconv(stream, m1[i].x_proj,    w1[i].xT,   ldx1,    di1, di1, 1);
    launch_wconv(stream, m1[i].dt_proj_w, w1[i].dtT,  di1,     dr1, 32,  1);
    launch_wconv(stream, m1[i].out_proj,  w1[i].outT, D1,      di1, di1, 1);
    launch_wconv(stream, m2[i].in_proj,   w2[i].inT,  2 * di2, D2,  D2,  1);
    launch_wconv(stream, m2[i].x_proj,    w2[i].xT,   ldx2,    di2, di2, 1);
    launch_wconv(stream, m2[i].dt_proj_w, w2[i].dtT,  di2,     dr2, 32,  1);
    launch_wconv(stream, m2[i].out_proj,  w2[i].outT, D2,      di2, di2, 1);
  }
  launch_wconv(stream, cls_w1, cw1T, 128, 256, 256, 1);
  launch_wconv(stream, cls_w2, cw2T, NCLS, 128, 128, 1);

  // ---- stage 1: patch conv (implicit WMMA GEMM) -> xp_c (B,64,196) ----
  patch_gemm_k<<<196, 128, 0, stream>>>(x_cwt, convwT, conv_b, xp_c);

  // ---- stage 2: projection + GELU, then LayerNorm -> xln (B,64,256) ----
  launch_acvt(stream, xp_c, xph, 1024, 196, 196, 224);
  launch_gemm(stream, xph, projT, xg, 1024, 224, 256, 256, proj_b, nullptr, 1);
  layernorm_k<<<(1024 + 7) / 8, 256, 0, stream>>>(xg, xln, ln_g, ln_b, 1024, 256);

  // ---- mamba block runner ----
  auto mblock = [&](const MambaP& p, const MambaWs& w, int Bn, int L, int D, int di,
                    int dr, int ldx, const float* xin, float* out) {
    int M = Bn * L, tot = M * di;
    launch_acvt(stream, xin, ah_x, M, D, D, D);
    launch_gemm(stream, ah_x, w.inT, xz, M, D, 2 * di, 2 * di, nullptr, nullptr, 0);
    dwconv_silu_k<<<(tot + 255) / 256, 256, 0, stream>>>(xz, p.conv_w, p.conv_b, xc, xch,
                                                         Bn, L, di, 2 * di);
    launch_gemm(stream, xch, w.xT, xdbl, M, di, ldx, ldx, nullptr, nullptr, 0);
    launch_acvt(stream, xdbl, dth, M, dr, ldx, 32);
    launch_gemm(stream, dth, w.dtT, dtb, M, 32, di, di, p.dt_proj_b, nullptr, 3);
    int nsc = Bn * di;
    scan_k<<<(nsc + 255) / 256, 256, 0, stream>>>(xc, dtb, xdbl, ldx, dr, p.A_log, p.D,
                                                  ybuf, Bn, L, di);
    gate_k<<<(tot + 255) / 256, 256, 0, stream>>>(ybuf, xz, yh, tot, di, 2 * di);
    launch_gemm(stream, yh, w.outT, out, M, di, D, D, nullptr, xin, 0);
  };
  auto runChain = [&](const MambaP* ps, const MambaWs* ws2, int Bn, int L, int D, int di,
                      int dr, int ldx, const float* xin, float* b0, float* b1) -> const float* {
    mblock(ps[0], ws2[0], Bn, L, D, di, dr, ldx, xin, b0);
    mblock(ps[1], ws2[1], Bn, L, D, di, dr, ldx, b0, b1);
    return b1;
  };

  const int NEi = (int)NE;
  // ---- mamba1: x1 = run(xln) + flip(run(flip(xln))) ----
  const float* xA = runChain(m1, w1, BB, L1, D1, di1, dr1, ldx1, xln, rbuf[0], rbuf[1]);
  flip_k<<<(NEi + 255) / 256, 256, 0, stream>>>(xln, fbuf, BB, L1, D1);
  const float* xBr = runChain(m1, w1, BB, L1, D1, di1, dr1, ldx1, fbuf, rbuf[2], rbuf[3]);
  flip_k<<<(NEi + 255) / 256, 256, 0, stream>>>(xBr, fbuf2, BB, L1, D1);
  add_k<<<(NEi + 255) / 256, 256, 0, stream>>>(xA, fbuf2, x1, NEi);

  // ---- mamba2 on transposed (B,256,64) ----
  transpose_k<<<(NEi + 255) / 256, 256, 0, stream>>>(x1, xpm, BB, L1, D1);           // -> (B,256,64)
  const float* r2a = runChain(m2, w2, BB, L2, D2, di2, dr2, ldx2, xpm, rbuf[4], rbuf[5]);
  flip_k<<<(NEi + 255) / 256, 256, 0, stream>>>(xpm, xpmf, BB, L2, D2);
  const float* r2b = runChain(m2, w2, BB, L2, D2, di2, dr2, ldx2, xpmf, rbuf[6], rbuf[7]);
  transpose_k<<<(NEi + 255) / 256, 256, 0, stream>>>(r2a, ta, BB, L2, D2);           // -> (B,64,256)
  transpose_k<<<(NEi + 255) / 256, 256, 0, stream>>>(r2b, tb, BB, L2, D2);
  flip_k<<<(NEi + 255) / 256, 256, 0, stream>>>(tb, tb2, BB, L1, D1);                // flip along 64
  add_k<<<(NEi + 255) / 256, 256, 0, stream>>>(ta, tb2, x2, NEi);

  // ---- pooling + classifier ----
  meanadd_k<<<(BB * PS + 255) / 256, 256, 0, stream>>>(x2, x1, x3, BB, L1, D1);
  launch_acvt(stream, x3, x3h, 16, 256, 256, 256);
  launch_gemm(stream, x3h, cw1T, t1, 16, 256, 128, 128, cls_b1, nullptr, 0);
  launch_acvt(stream, t1, t1h, 16, 128, 128, 128);
  launch_gemm(stream, t1h, cw2T, (float*)d_out, 16, 128, NCLS, NCLS, cls_b2, nullptr, 0);
}
